// TalkingHeadAttn_76115410420296
// MI455X (gfx1250) — matrix operand order repaired and checked by reading, not compiled, not run on hardware
//
#include <hip/hip_runtime.h>

// ---------------------------------------------------------------------------
// Types for CDNA5 WMMA (wave32): v_wmma_f32_16x16x32_bf16
// ---------------------------------------------------------------------------
typedef __attribute__((ext_vector_type(16))) __bf16          v16bf;
typedef __attribute__((ext_vector_type(16))) unsigned short  v16u;
typedef __attribute__((ext_vector_type(8)))  float           v8f;
typedef __attribute__((ext_vector_type(4)))  unsigned int    v4u;
typedef __attribute__((ext_vector_type(4)))  int             v4i;
typedef __attribute__((ext_vector_type(8)))  int             v8i;

#define BATCH 8
#define SEQ   1024
#define CDIM  768
#define HEADS 12
#define HDIM  64
#define ROWS  (BATCH * SEQ)          // 8192
#define QKVN  (3 * CDIM)             // 2304

#if __has_builtin(__builtin_amdgcn_tensor_load_to_lds)
#define HAVE_TDM 1
#else
#define HAVE_TDM 0
#endif

__device__ __forceinline__ unsigned short f2bf(float f) {
  union { float f; unsigned u; } v; v.f = f;
  unsigned u = v.u;
  return (unsigned short)((u + 0x7FFFu + ((u >> 16) & 1u)) >> 16); // RNE
}

__device__ __forceinline__ v16u pack2(v4u lo, v4u hi) {
  union { v4u q[2]; v16u s; } u;
  u.q[0] = lo; u.q[1] = hi;
  return u.s;
}

__device__ __forceinline__ v8f wmma_bf16(v16u a, v16u b, v8f c) {
  return __builtin_amdgcn_wmma_f32_16x16x32_bf16(
      /*neg_a=*/false, __builtin_bit_cast(v16bf, a),
      /*neg_b=*/false, __builtin_bit_cast(v16bf, b),
      /*c_mod=*/(short)0, c, /*reuse_a=*/false, /*reuse_b=*/false);
}

// A fragment (16x32, M x K): lane m holds row m; two contiguous 8-element K
// chunks at offsets h8 and 16+h8.  Two 128-bit loads (16B aligned by layout).
__device__ __forceinline__ v16u load_a_frag(const unsigned short* base, int ld, int lane) {
  int m  = lane & 15;
  int h8 = (lane >> 4) << 3;
  const unsigned short* p0 = base + (size_t)m * ld + h8;
  return pack2(*(const v4u*)p0, *(const v4u*)(p0 + 16));
}

// B fragment (32x16, K x N) from a row-major [K, N] matrix: lane n = column,
// K elements h8..h8+7 and 16+h8..16+h8+7 (strided gather).  base at [k0, n0].
__device__ __forceinline__ v16u load_b_frag(const unsigned short* base, int ld, int lane) {
  const unsigned short* p = base + (lane & 15);
  int h8 = (lane >> 4) << 3;
  v16u r;
#pragma unroll
  for (int e = 0; e < 8; ++e) {
    r[e]     = p[(size_t)(h8 + e) * ld];
    r[8 + e] = p[(size_t)(16 + h8 + e) * ld];
  }
  return r;
}

// ---------------------------------------------------------------------------
// fp32 -> bf16 convert (straight) and convert+transpose (for weights / K)
// ---------------------------------------------------------------------------
__global__ void k_cvt(const float* __restrict__ src, unsigned short* __restrict__ dst, int n) {
  int i = blockIdx.x * blockDim.x + threadIdx.x;
  if (i < n) dst[i] = f2bf(src[i]);
}

// src row-major [R, C]  ->  dst row-major [C, R]  (dst[k*R + n] = src[n*C + k])
__global__ void k_cvt_t(const float* __restrict__ src, unsigned short* __restrict__ dst,
                        int R, int C) {
  int i = blockIdx.x * blockDim.x + threadIdx.x;
  if (i < R * C) {
    int n = i / C, k = i % C;
    dst[(size_t)k * R + n] = f2bf(src[i]);
  }
}

// ---------------------------------------------------------------------------
// QKV GEMM: x[8192,768]bf16 x qkv_wT[768,2304]bf16 -> q(*scale)[B,H,N,D],
// kT[B,H,D,N], v[B,H,N,D], all bf16.
// grid (144, 64) x 256 thr (8 waves); each wave owns one 16x16 output tile.
// ---------------------------------------------------------------------------
__global__ void k_qkv_gemm(const unsigned short* __restrict__ xb,
                           const unsigned short* __restrict__ wqt,  // [768,2304]
                           unsigned short* __restrict__ qb,
                           unsigned short* __restrict__ kt,
                           unsigned short* __restrict__ vb) {
  int lane = threadIdx.x & 31;
  int wave = threadIdx.x >> 5;
  int n0 = blockIdx.x * 16;
  int m0 = (blockIdx.y * 8 + wave) * 16;
  v8f c = {};
  for (int k0 = 0; k0 < CDIM; k0 += 32) {
    __builtin_prefetch(xb + (size_t)m0 * CDIM + k0 + 64, 0, 1);
    v16u a = load_a_frag(xb + (size_t)m0 * CDIM + k0, CDIM, lane);
    v16u b = load_b_frag(wqt + (size_t)k0 * QKVN + n0, QKVN, lane);
    c = wmma_bf16(a, b, c);
  }
  int nn    = n0 + (lane & 15);
  int which = nn / CDIM;                 // 0=q 1=k 2=v (uniform per tile)
  int crem  = nn % CDIM;
  int h = crem >> 6, d = crem & 63;
  float scale = (which == 0) ? 0.125f : 1.0f;   // HEAD_DIM^-0.5
  int rbase = (lane >> 4) << 3;
#pragma unroll
  for (int r = 0; r < 8; ++r) {
    int m  = m0 + r + rbase;
    int bi = m >> 10, i = m & 1023;
    unsigned short val = f2bf(c[r] * scale);
    if (which == 0)      qb[((bi * HEADS + h) * SEQ + i) * HDIM + d] = val;
    else if (which == 1) kt[((bi * HEADS + h) * HDIM + d) * SEQ + i] = val;
    else                 vb[((bi * HEADS + h) * SEQ + i) * HDIM + d] = val;
  }
}

// ---------------------------------------------------------------------------
// Logits + pre-softmax talking-heads mix.
// Block = (i-block of 16 rows, batch b); 12 waves; wave h computes S_h = q_h k_h^T
// tiles via WMMA (B fragments are strided gathers from kT, stride SEQ), exchanges
// through LDS, then writes M[g] = sum_h wl[g,h] S_h + bl[g] into soft_attn.
// ---------------------------------------------------------------------------
__global__ void k_logits(const unsigned short* __restrict__ qb,
                         const unsigned short* __restrict__ kt,
                         const float* __restrict__ wl,
                         const float* __restrict__ bl,
                         float* __restrict__ attn) {
  __shared__ float Sld[HEADS][256];
  int lane = threadIdx.x & 31;
  int h    = threadIdx.x >> 5;       // wave id == head (and mix output head g)
  int i0   = blockIdx.x * 16;
  int b    = blockIdx.y;

  const unsigned short* qbase = qb + ((size_t)(b * HEADS + h) * SEQ + i0) * HDIM;
  v16u aq0 = load_a_frag(qbase,      HDIM, lane);   // K(d) = 0..31
  v16u aq1 = load_a_frag(qbase + 32, HDIM, lane);   // K(d) = 32..63

  const unsigned short* ktb = kt + (size_t)(b * HEADS + h) * HDIM * SEQ;  // [d][j]

  float wrow[HEADS];
#pragma unroll
  for (int t = 0; t < HEADS; ++t) wrow[t] = wl[h * HEADS + t];
  float bias = bl[h];

  float* out = attn + ((size_t)(b * HEADS + h) * SEQ + i0) * SEQ;

  for (int jt = 0; jt < SEQ / 16; ++jt) {
    __builtin_prefetch(ktb + jt * 16 + 16, 0, 1);
    v16u bk0 = load_b_frag(ktb + jt * 16,                  SEQ, lane);  // d = 0..31
    v16u bk1 = load_b_frag(ktb + (size_t)32 * SEQ + jt * 16, SEQ, lane); // d = 32..63
    v8f s = {};
    s = wmma_bf16(aq0, bk0, s);
    s = wmma_bf16(aq1, bk1, s);
#pragma unroll
    for (int r = 0; r < 8; ++r) Sld[h][r * 32 + lane] = s[r];
    __syncthreads();
#pragma unroll
    for (int r = 0; r < 8; ++r) {
      float acc = bias;
#pragma unroll
      for (int hh = 0; hh < HEADS; ++hh) acc += wrow[hh] * Sld[hh][r * 32 + lane];
      int row = r + ((lane >> 4) << 3);
      out[(size_t)row * SEQ + jt * 16 + (lane & 15)] = acc;
    }
    __syncthreads();
  }
}

// ---------------------------------------------------------------------------
// Row softmax in place over the last axis (1024). One block per row.
// ---------------------------------------------------------------------------
__global__ void k_softmax(float* __restrict__ attn) {
  __shared__ float red[256];
  float* p = attn + (size_t)blockIdx.x * SEQ;
  int t = threadIdx.x;
  float v0 = p[t], v1 = p[t + 256], v2 = p[t + 512], v3 = p[t + 768];
  float m = fmaxf(fmaxf(v0, v1), fmaxf(v2, v3));
  red[t] = m; __syncthreads();
  for (int s = 128; s > 0; s >>= 1) { if (t < s) red[t] = fmaxf(red[t], red[t + s]); __syncthreads(); }
  m = red[0]; __syncthreads();
  v0 = __expf(v0 - m); v1 = __expf(v1 - m); v2 = __expf(v2 - m); v3 = __expf(v3 - m);
  red[t] = v0 + v1 + v2 + v3; __syncthreads();
  for (int s = 128; s > 0; s >>= 1) { if (t < s) red[t] += red[t + s]; __syncthreads(); }
  float inv = 1.0f / red[0];
  p[t] = v0 * inv; p[t + 256] = v1 * inv; p[t + 512] = v2 * inv; p[t + 768] = v3 * inv;
}

// ---------------------------------------------------------------------------
// TDM: stage one [12 heads x 16 rows x 32 cols] f32 tile of the attention map
// into LDS.  3-D descriptor per ISA ch.8: x stride 1 (j), y stride 1024 (i),
// z stride 1024*1024 (head).  LDS receives [z][y][x] contiguous.
// ---------------------------------------------------------------------------
#if HAVE_TDM
__device__ __forceinline__ void tdm_load_tile(const float* gptr, unsigned lds_off) {
  unsigned long long ga = (unsigned long long)(uintptr_t)gptr;
  v4u g0;
  g0[0] = 1u;                                             // count=1, user mode
  g0[1] = lds_off;                                        // lds_addr (bytes)
  g0[2] = (unsigned)(ga & 0xFFFFFFFFu);                   // global_addr lo
  g0[3] = (unsigned)((ga >> 32) & 0x01FFFFFFu) | (2u << 30); // addr hi | type=2
  v8i g1;
  g1[0] = (int)(2u << 16);                                // data_size=4B, mask=0
  g1[1] = (int)((1024u & 0xFFFFu) << 16);                 // tensor_dim0 lo16
  g1[2] = (int)((1024u >> 16) | ((0x100000u & 0xFFFFu) << 16)); // d0 hi | d1 lo
  g1[3] = (int)((0x100000u >> 16) | (32u << 16));         // d1 hi | tile_dim0=32
  g1[4] = (int)(16u | (12u << 16));                       // tile_dim1=16 | tile_dim2=12
  g1[5] = (int)1024;                                      // tensor_dim0_stride lo32
  g1[6] = (int)((1048576u & 0xFFFFu) << 16);              // d0s hi=0 | dim1_stride lo16
  g1[7] = (int)(1048576u >> 16);                          // dim1_stride [47:16]
  v4i g2; g2[0] = 64; g2[1] = 0; g2[2] = 0; g2[3] = 0;    // tensor_dim2 >= 12
  v4i g3 = {0, 0, 0, 0};
#if __clang_major__ >= 23
  v8i gz = {0, 0, 0, 0, 0, 0, 0, 0};
  __builtin_amdgcn_tensor_load_to_lds(g0, g1, g2, g3, gz, 0);
#else
  __builtin_amdgcn_tensor_load_to_lds(g0, g1, g2, g3, 0);
#endif
}
#endif

// ---------------------------------------------------------------------------
// Post-softmax mix + AV: out[g] = (sum_h ww[g,h] P[h] + bw[g]) @ v[g].
// Block = (i-block 16, b); 12 waves.  TDM double-buffers the 12-head P tiles
// into LDS (overlapping DMA with WMMA); wave g mixes into a bf16 A fragment,
// then 4 WMMAs against v[g] per 32-wide j chunk.
// ---------------------------------------------------------------------------
__global__ void k_av(const float* __restrict__ attn,
                     const unsigned short* __restrict__ vb,
                     const float* __restrict__ ww,
                     const float* __restrict__ bw,
                     unsigned short* __restrict__ aob) {
  __shared__ float Pld[2][HEADS][16][32];
  int lane = threadIdx.x & 31;
  int g    = threadIdx.x >> 5;
  int i0   = blockIdx.x * 16;
  int b    = blockIdx.y;

  float wrow[HEADS];
#pragma unroll
  for (int t = 0; t < HEADS; ++t) wrow[t] = ww[g * HEADS + t];
  float bias = bw[g];

  v8f acc0 = {}, acc1 = {}, acc2 = {}, acc3 = {};
  int m  = lane & 15;
  int h8 = (lane >> 4) << 3;

  const float* gbase = attn + ((size_t)b * HEADS * SEQ + i0) * SEQ;  // head 0 tile base

#if HAVE_TDM
  if (g == 0) tdm_load_tile(gbase, (unsigned)(uintptr_t)&Pld[0][0][0][0]);
#endif

  for (int j0 = 0; j0 < SEQ; j0 += 32) {
    int cur = (j0 >> 5) & 1;
#if HAVE_TDM
    if (g == 0) {
      if (j0 + 32 < SEQ) {
        tdm_load_tile(gbase + j0 + 32, (unsigned)(uintptr_t)&Pld[cur ^ 1][0][0][0]);
        __builtin_amdgcn_s_wait_tensorcnt(1);   // chunk j0 complete (in-order)
      } else {
        __builtin_amdgcn_s_wait_tensorcnt(0);
      }
    }
#else
    { // fallback: wave g stages its own head's 16x32 tile, float4 loads
      int row = lane >> 1;
      int c0  = (lane & 1) * 16;
      const float* src = attn + ((size_t)(b * HEADS + g) * SEQ + i0 + row) * SEQ + j0 + c0;
      *(float4*)&Pld[cur][g][row][c0 + 0]  = *(const float4*)(src + 0);
      *(float4*)&Pld[cur][g][row][c0 + 4]  = *(const float4*)(src + 4);
      *(float4*)&Pld[cur][g][row][c0 + 8]  = *(const float4*)(src + 8);
      *(float4*)&Pld[cur][g][row][c0 + 12] = *(const float4*)(src + 12);
    }
#endif
    __syncthreads();
    // mixed bf16 A fragment for output head g
    v16u a;
#pragma unroll
    for (int e = 0; e < 8; ++e) {
      int k1 = h8 + e, k2 = 16 + h8 + e;
      float fa = bias, fb = bias;
#pragma unroll
      for (int hh = 0; hh < HEADS; ++hh) {
        fa += wrow[hh] * Pld[cur][hh][m][k1];
        fb += wrow[hh] * Pld[cur][hh][m][k2];
      }
      a[e] = f2bf(fa); a[8 + e] = f2bf(fb);
    }
    const unsigned short* vbase = vb + ((size_t)(b * HEADS + g) * SEQ + j0) * HDIM;
    acc0 = wmma_bf16(a, load_b_frag(vbase +  0, HDIM, lane), acc0);
    acc1 = wmma_bf16(a, load_b_frag(vbase + 16, HDIM, lane), acc1);
    acc2 = wmma_bf16(a, load_b_frag(vbase + 32, HDIM, lane), acc2);
    acc3 = wmma_bf16(a, load_b_frag(vbase + 48, HDIM, lane), acc3);
    __syncthreads();
  }

  unsigned short* dst = aob + (size_t)(b * SEQ + i0) * CDIM + g * HDIM;
  int nlo = lane & 15, rbase = (lane >> 4) << 3;
#pragma unroll
  for (int r = 0; r < 8; ++r) {
    size_t off = (size_t)(r + rbase) * CDIM;
    dst[off +  0 + nlo] = f2bf(acc0[r]);
    dst[off + 16 + nlo] = f2bf(acc1[r]);
    dst[off + 32 + nlo] = f2bf(acc2[r]);
    dst[off + 48 + nlo] = f2bf(acc3[r]);
  }
}

// ---------------------------------------------------------------------------
// Final projection: out = ao @ proj_w^T + proj_b  (fp32 output)
// ---------------------------------------------------------------------------
__global__ void k_proj(const unsigned short* __restrict__ aob,
                       const unsigned short* __restrict__ wpt,   // [768,768] transposed
                       const float* __restrict__ pbias,
                       float* __restrict__ out) {
  int lane = threadIdx.x & 31;
  int wave = threadIdx.x >> 5;
  int n0 = blockIdx.x * 16;
  int m0 = (blockIdx.y * 8 + wave) * 16;
  v8f c = {};
  for (int k0 = 0; k0 < CDIM; k0 += 32) {
    __builtin_prefetch(aob + (size_t)m0 * CDIM + k0 + 64, 0, 1);
    v16u a = load_a_frag(aob + (size_t)m0 * CDIM + k0, CDIM, lane);
    v16u b = load_b_frag(wpt + (size_t)k0 * CDIM + n0, CDIM, lane);
    c = wmma_bf16(a, b, c);
  }
  int nn = n0 + (lane & 15);
  float bias = pbias[nn];
  int rbase = (lane >> 4) << 3;
#pragma unroll
  for (int r = 0; r < 8; ++r) {
    int mrow = m0 + r + rbase;
    out[(size_t)mrow * CDIM + nn] = c[r] + bias;
  }
}

// ---------------------------------------------------------------------------
extern "C" void kernel_launch(void* const* d_in, const int* in_sizes, int n_in,
                              void* d_out, int out_size, void* d_ws, size_t ws_size,
                              hipStream_t stream) {
  const float* x      = (const float*)d_in[0];
  const float* qkv_w  = (const float*)d_in[1];
  const float* proj_w = (const float*)d_in[2];
  const float* proj_b = (const float*)d_in[3];
  const float* wl     = (const float*)d_in[4];
  const float* bl     = (const float*)d_in[5];
  const float* ww     = (const float*)d_in[6];
  const float* bw     = (const float*)d_in[7];

  float* out       = (float*)d_out;
  float* soft_attn = out + (size_t)ROWS * CDIM;   // second tuple output

  // workspace layout (bf16 elements; every region offset is 16B aligned)
  unsigned short* xb  = (unsigned short*)d_ws;
  unsigned short* wqt = xb  + (size_t)ROWS * CDIM;                  // [768,2304]
  unsigned short* wpt = wqt + (size_t)QKVN * CDIM;                  // [768,768]
  unsigned short* qb  = wpt + (size_t)CDIM * CDIM;                  // [B,H,N,D]
  unsigned short* kt  = qb  + (size_t)BATCH * HEADS * SEQ * HDIM;   // [B,H,D,N]
  unsigned short* vb  = kt  + (size_t)BATCH * HEADS * SEQ * HDIM;   // [B,H,N,D]
  unsigned short* aob = vb  + (size_t)BATCH * HEADS * SEQ * HDIM;   // [B*N,C]

  const int n_x = ROWS * CDIM, n_wq = QKVN * CDIM, n_wp = CDIM * CDIM;
  k_cvt  <<<(n_x  + 255) / 256, 256, 0, stream>>>(x, xb, n_x);
  k_cvt_t<<<(n_wq + 255) / 256, 256, 0, stream>>>(qkv_w,  wqt, QKVN, CDIM);
  k_cvt_t<<<(n_wp + 255) / 256, 256, 0, stream>>>(proj_w, wpt, CDIM, CDIM);

  k_qkv_gemm<<<dim3(QKVN / 16, ROWS / 128), 256, 0, stream>>>(xb, wqt, qb, kt, vb);
  k_logits  <<<dim3(SEQ / 16, BATCH), HEADS * 32, 0, stream>>>(qb, kt, wl, bl, soft_attn);
  k_softmax <<<BATCH * HEADS * SEQ, 256, 0, stream>>>(soft_attn);
  k_av      <<<dim3(SEQ / 16, BATCH), HEADS * 32, 0, stream>>>(soft_attn, vb, ww, bw, aob);
  k_proj    <<<dim3(CDIM / 16, ROWS / 128), 256, 0, stream>>>(aob, wpt, proj_b, out);
}
